// SelfAttention_Layer_69045894250572
// MI455X (gfx1250) — matrix operand order repaired and checked by reading, not compiled, not run on hardware
//
#include <hip/hip_runtime.h>

#define DEV __device__ __forceinline__

typedef __bf16 bf16;
typedef __attribute__((ext_vector_type(8)))  bf16  bf16x8;
typedef __attribute__((ext_vector_type(16))) bf16  bf16x16;
typedef __attribute__((ext_vector_type(8)))  float v8f;

#define NEG_INF_F (-4294967295.0f)  /* -(2^32)+1, matches reference */

#if __has_builtin(__builtin_amdgcn_tensor_load_to_lds) && __has_builtin(__builtin_amdgcn_s_wait_tensorcnt)
#define USE_TDM 1
typedef unsigned int u32x4 __attribute__((ext_vector_type(4)));
typedef int          i32x8 __attribute__((ext_vector_type(8)));
typedef int          i32x4 __attribute__((ext_vector_type(4)));
#endif

DEV bf16 f2bf(float f) { return (bf16)f; }   // native v_cvt on gfx1250

DEV v8f v8f_zero() {
    v8f r;
    #pragma unroll
    for (int i = 0; i < 8; ++i) r[i] = 0.f;
    return r;
}

// sinusoidal positional encoding value for (s, d), d_model = 256
DEV float pe_val(int s, int d) {
    float expo = (float)(d & ~1) * (1.0f / 256.0f);
    float rate = __expf(-expo * 9.210340371976184f);   // ln(10000)
    float ang  = (float)s * rate;
    return (d & 1) ? __cosf(ang) : __sinf(ang);
}

union AU { bf16x16 v; bf16x8 h[2]; };

#define WMMA_BF16(A, B, C) __builtin_amdgcn_wmma_f32_16x16x32_bf16( \
        false, (A), false, (B), (short)0, (C), false, false)

// ---------------------------------------------------------------------------
// proj: out[row, :] = relu((X[row,:] + PE[row%512,:]) @ W), out in bf16
// grid.x = (B*S)/128, block = 256 (8 waves, 16 rows each)
// ---------------------------------------------------------------------------
__global__ __launch_bounds__(256)
void proj_kernel(const float* __restrict__ X, const float* __restrict__ W,
                 bf16* __restrict__ out)
{
    __shared__ alignas(16) bf16 Wt[256 * 32];   // [col][k] transposed K-chunk

    const int tid  = threadIdx.x;
    const int wave = tid >> 5;
    const int lane = tid & 31;
    const int l16  = lane & 15;
    const bool hi  = lane >= 16;
    const int rowbase = blockIdx.x * 128 + wave * 16;
    const int row = rowbase + l16;        // A-matrix row for this lane
    const int s   = row & 511;            // row % S
    const int off0 = hi ? 8 : 0;
    const int boff = hi ? 8 : 0;

    v8f acc[16];
    #pragma unroll
    for (int t = 0; t < 16; ++t) acc[t] = v8f_zero();

    for (int k0 = 0; k0 < 256; k0 += 32) {
        __syncthreads();
        // stage W chunk transposed: Wt[c*32 + k] = bf16(W[(k0+k)*256 + c])
        for (int e = tid; e < 32 * 256; e += 256) {
            int k = e >> 8, c = e & 255;
            Wt[c * 32 + k] = f2bf(W[(k0 + k) * 256 + c]);
        }
        __syncthreads();

        // A fragment: row, cols k0+{off0..off0+7} and k0+{off0+16..off0+23}
        float xv[16];
        const float* xr = X + (size_t)row * 256;
        if (k0 + 32 < 256) __builtin_prefetch(xr + k0 + 32, 0, 3);
        #pragma unroll
        for (int h = 0; h < 2; ++h) {
            int cbase = k0 + off0 + h * 16;
            float4 a = *(const float4*)(xr + cbase);
            float4 b = *(const float4*)(xr + cbase + 4);
            xv[h*8+0]=a.x; xv[h*8+1]=a.y; xv[h*8+2]=a.z; xv[h*8+3]=a.w;
            xv[h*8+4]=b.x; xv[h*8+5]=b.y; xv[h*8+6]=b.z; xv[h*8+7]=b.w;
            #pragma unroll
            for (int i = 0; i < 8; ++i) xv[h*8+i] += pe_val(s, cbase + i);
        }
        AU afr;
        #pragma unroll
        for (int i = 0; i < 8; ++i) { afr.h[0][i] = f2bf(xv[i]); afr.h[1][i] = f2bf(xv[8+i]); }

        // 16 N-tiles, software-pipelined B-fragment loads from LDS
        AU bcur, bnxt;
        {
            const bf16* wp = &Wt[l16 * 32 + boff];
            bcur.h[0] = *(const bf16x8*)(wp);
            bcur.h[1] = *(const bf16x8*)(wp + 16);
        }
        #pragma unroll
        for (int t = 0; t < 16; ++t) {
            if (t < 15) {
                const bf16* wp = &Wt[((t + 1) * 16 + l16) * 32 + boff];
                bnxt.h[0] = *(const bf16x8*)(wp);
                bnxt.h[1] = *(const bf16x8*)(wp + 16);
            }
            acc[t] = WMMA_BF16(afr.v, bcur.v, acc[t]);
            bcur = bnxt;
        }
    }

    // relu + bf16 store (C layout: VGPR j -> row j / j+8 by lane half)
    #pragma unroll
    for (int t = 0; t < 16; ++t) {
        int c = t * 16 + l16;
        #pragma unroll
        for (int j = 0; j < 8; ++j) {
            int r = rowbase + j + (hi ? 8 : 0);
            float v = acc[t][j];
            out[(size_t)r * 256 + c] = f2bf(v > 0.f ? v : 0.f);
        }
    }
}

// ---------------------------------------------------------------------------
// attention: flash-style, wave = 16 queries x 256 dims, 32-key blocks
// grid = (B, S/128), block = 256
// ---------------------------------------------------------------------------
__global__ __launch_bounds__(256)
void attn_kernel(const bf16* __restrict__ qp, const bf16* __restrict__ kp,
                 const float* __restrict__ v, const int* __restrict__ mask,
                 float* __restrict__ out)
{
    __shared__ alignas(16) bf16 Kblk[32 * 256];   // [key][d] row-major bf16
    __shared__ alignas(16) bf16 Vt[256 * 32];     // [d][key] transposed bf16
    __shared__ alignas(16) bf16 Pbuf[8][16 * 32]; // per-wave P transpose buffer

    const int b    = blockIdx.x;
    const int tid  = threadIdx.x;
    const int wave = tid >> 5;
    const int lane = tid & 31;
    const int l16  = lane & 15;
    const bool hi  = lane >= 16;
    const int off0 = hi ? 8 : 0;
    const int boff = hi ? 8 : 0;
    const int qrow = b * 512 + blockIdx.y * 128 + wave * 16 + l16;

    // preload Q A-fragments for all 8 K-steps over D
    AU qfr[8];
    const bf16* qr = qp + (size_t)qrow * 256;
    #pragma unroll
    for (int k = 0; k < 8; ++k) {
        qfr[k].h[0] = *(const bf16x8*)(qr + k * 32 + off0);
        qfr[k].h[1] = *(const bf16x8*)(qr + k * 32 + off0 + 16);
    }

    v8f acc[16];
    #pragma unroll
    for (int t = 0; t < 16; ++t) acc[t] = v8f_zero();
    float m[8], lsum[8];
    #pragma unroll
    for (int j = 0; j < 8; ++j) { m[j] = -3.0e38f; lsum[j] = 0.f; }

    const float* vbase = v  + (size_t)b * 512 * 256;
    const bf16*  kbase = kp + (size_t)b * 512 * 256;

    for (int kb = 0; kb < 512; kb += 32) {
        __syncthreads();

        // ---- stage K block: TDM tensor DMA if available, else b128 copy ----
#ifdef USE_TDM
        if (wave == 0) {
            size_t   ga = (size_t)(const void*)(kbase + (size_t)kb * 256);
            unsigned la = (unsigned)(size_t)(void*)&Kblk[0];
            u32x4 g0;
            g0[0] = 1u;                                        // count=1
            g0[1] = la;                                        // lds_addr
            g0[2] = (unsigned)(ga & 0xFFFFFFFFu);              // global_addr lo
            g0[3] = (unsigned)((ga >> 32) & 0x01FFFFFFu) | 0x80000000u; // hi | type=2
            i32x8 g1;
            g1[0] = 0x00010000;          // workgroup_mask=0, data_size=1 (2B)
            g1[1] = (int)(256u << 16);   // tensor_dim0 = 256 (lo16<<16)
            g1[2] = (int)(512u << 16);   // tensor_dim1 = 512 (lo16<<16)
            g1[3] = (int)(256u << 16);   // tile_dim0 = 256
            g1[4] = 32;                  // tile_dim1 = 32, tile_dim2 = 0
            g1[5] = 256;                 // tensor_dim0_stride lo32
            g1[6] = (int)(256u << 16);   // t0s hi=0 | tensor_dim1_stride lo16
            g1[7] = 0;
            i32x4 gz4; gz4[0] = gz4[1] = gz4[2] = gz4[3] = 0;
            i32x8 gz8;
            #pragma unroll
            for (int i = 0; i < 8; ++i) gz8[i] = 0;
            __builtin_amdgcn_tensor_load_to_lds(g0, g1, gz4, gz4, gz8, 0);
            __builtin_amdgcn_s_wait_tensorcnt(0);
        }
#else
        {
            const bf16* src = kbase + (size_t)kb * 256;
            #pragma unroll
            for (int i = 0; i < 4; ++i) {
                int o = (tid + i * 256) * 8;
                *(bf16x8*)&Kblk[o] = *(const bf16x8*)(src + o);
            }
        }
#endif
        // prefetch next K/V blocks while we convert this one
        if (kb + 32 < 512) {
            __builtin_prefetch(kbase + (size_t)(kb + 32) * 256 + tid * 32, 0, 3);
            __builtin_prefetch(vbase + (size_t)(kb + 32) * 256 + tid * 16, 0, 3);
            __builtin_prefetch(vbase + (size_t)(kb + 32) * 256 + (tid + 256) * 16, 0, 3);
        }
        // stage V block transposed with f32->bf16 convert (tid indexes d)
        #pragma unroll
        for (int kk = 0; kk < 32; kk += 8) {
            bf16x8 pk;
            #pragma unroll
            for (int j = 0; j < 8; ++j)
                pk[j] = f2bf(vbase[(size_t)(kb + kk + j) * 256 + tid]);
            *(bf16x8*)&Vt[tid * 32 + kk] = pk;
        }
        __syncthreads();

        // S = Q K^T  (two 16-key subtiles x 8 K-steps, pipelined LDS loads)
        v8f st[2]; st[0] = v8f_zero(); st[1] = v8f_zero();
        {
            AU bcur, bnxt;
            {
                const bf16* p = &Kblk[l16 * 256 + boff];
                bcur.h[0] = *(const bf16x8*)(p);
                bcur.h[1] = *(const bf16x8*)(p + 16);
            }
            #pragma unroll
            for (int i = 0; i < 16; ++i) {
                int sb = i >> 3, k = i & 7;
                if (i < 15) {
                    int i2 = i + 1, sb2 = i2 >> 3, k2 = i2 & 7;
                    const bf16* p = &Kblk[(sb2 * 16 + l16) * 256 + k2 * 32 + boff];
                    bnxt.h[0] = *(const bf16x8*)(p);
                    bnxt.h[1] = *(const bf16x8*)(p + 16);
                }
                st[sb] = WMMA_BF16(qfr[k].v, bcur.v, st[sb]);
                bcur = bnxt;
            }
        }

        // scale + key mask (key index = lane column in C layout)
        #pragma unroll
        for (int sb = 0; sb < 2; ++sb) {
            int key = kb + sb * 16 + l16;
            bool mz = (mask[b * 512 + key] == 0);
            #pragma unroll
            for (int j = 0; j < 8; ++j) {
                float x = st[sb][j] * 0.0625f;       // 1/sqrt(256)
                st[sb][j] = mz ? NEG_INF_F : x;
            }
        }

        // online softmax: row max / exp / row sum (rows = VGPR idx per half)
        float alpha[8];
        #pragma unroll
        for (int j = 0; j < 8; ++j) {
            float mx = fmaxf(st[0][j], st[1][j]);
            #pragma unroll
            for (int o = 1; o < 16; o <<= 1) mx = fmaxf(mx, __shfl_xor(mx, o, 32));
            float mn = fmaxf(m[j], mx);
            alpha[j] = __expf(m[j] - mn);
            m[j] = mn;
        }
        #pragma unroll
        for (int j = 0; j < 8; ++j) {
            float p0 = __expf(st[0][j] - m[j]);
            float p1 = __expf(st[1][j] - m[j]);
            st[0][j] = p0; st[1][j] = p1;
            float s = p0 + p1;
            #pragma unroll
            for (int o = 1; o < 16; o <<= 1) s += __shfl_xor(s, o, 32);
            lsum[j] = lsum[j] * alpha[j] + s;
        }
        #pragma unroll
        for (int t = 0; t < 16; ++t) {
            #pragma unroll
            for (int j = 0; j < 8; ++j) acc[t][j] *= alpha[j];
        }

        // P: C layout -> A layout via per-wave LDS buffer
        bf16* pb = Pbuf[wave];
        #pragma unroll
        for (int sb = 0; sb < 2; ++sb) {
            int col = sb * 16 + l16;
            #pragma unroll
            for (int j = 0; j < 8; ++j)
                pb[(j + (hi ? 8 : 0)) * 32 + col] = f2bf(st[sb][j]);
        }
        AU pfr;
        {
            const bf16* pr = pb + l16 * 32 + off0;
            pfr.h[0] = *(const bf16x8*)(pr);
            pfr.h[1] = *(const bf16x8*)(pr + 16);
        }

        // O += P @ V   (16 N-tiles over D, pipelined B frags from Vt)
        {
            AU bcur, bnxt;
            {
                const bf16* vp = &Vt[l16 * 32 + boff];
                bcur.h[0] = *(const bf16x8*)(vp);
                bcur.h[1] = *(const bf16x8*)(vp + 16);
            }
            #pragma unroll
            for (int t = 0; t < 16; ++t) {
                if (t < 15) {
                    const bf16* vp = &Vt[((t + 1) * 16 + l16) * 32 + boff];
                    bnxt.h[0] = *(const bf16x8*)(vp);
                    bnxt.h[1] = *(const bf16x8*)(vp + 16);
                }
                acc[t] = WMMA_BF16(pfr.v, bcur.v, acc[t]);
                bcur = bnxt;
            }
        }
    }

    // finalize: out[b, d] += sum over this wave's 16 queries of acc/lsum, / 512 (mean over S)
    #pragma unroll
    for (int t = 0; t < 16; ++t) {
        float s = 0.f;
        #pragma unroll
        for (int j = 0; j < 8; ++j) s += acc[t][j] / lsum[j];
        s += __shfl_xor(s, 16, 32);                 // add the other 8 rows
        if (!hi) atomicAdd(&out[b * 256 + t * 16 + l16], s * (1.0f / 512.0f));
    }
}

__global__ void zero_kernel(float* __restrict__ out, int n) {
    int i = blockIdx.x * 256 + threadIdx.x;
    if (i < n) out[i] = 0.f;
}

extern "C" void kernel_launch(void* const* d_in, const int* in_sizes, int n_in,
                              void* d_out, int out_size, void* d_ws, size_t ws_size,
                              hipStream_t stream)
{
    (void)in_sizes; (void)n_in; (void)out_size; (void)ws_size;
    const float* q    = (const float*)d_in[0];
    const float* k    = (const float*)d_in[1];
    const float* v    = (const float*)d_in[2];
    const float* W    = (const float*)d_in[3];
    const int*   mask = (const int*)d_in[4];
    float* out = (float*)d_out;

    const size_t NTOK = (size_t)64 * 512 * 256;   // B*S*D
    bf16* qp = (bf16*)d_ws;
    bf16* kp = qp + NTOK;

    zero_kernel<<<64, 256, 0, stream>>>(out, 64 * 256);
    proj_kernel<<<256, 256, 0, stream>>>(q, W, qp);   // q projection
    proj_kernel<<<256, 256, 0, stream>>>(k, W, kp);   // k projection
    attn_kernel<<<dim3(64, 4), 256, 0, stream>>>(qp, kp, v, mask, out);
}
